// BERT_CRF_54279796686961
// MI455X (gfx1250) — matrix operand order repaired and checked by reading, not compiled, not run on hardware
//
#include <hip/hip_runtime.h>
#include <math.h>

// ---------------- problem constants ----------------
static constexpr int B_   = 8;
static constexpr int T_   = 128;
static constexpr int L_   = 12;
static constexpr int H_   = 768;
static constexpr int NH_  = 12;
static constexpr int F_   = 3072;
static constexpr int KT   = 5;       // CRF tags
static constexpr int DH_  = 64;      // head dim
static constexpr int QKV_ = 3 * H_;  // 2304
static constexpr int NTOK = B_ * T_; // 1024
static constexpr int STARTTAG = 3;

// ---------------- types ----------------
typedef __bf16 bf16;
typedef __attribute__((ext_vector_type(16))) __bf16 v16bf;
typedef __attribute__((ext_vector_type(8)))  __bf16 v8bf;
typedef __attribute__((ext_vector_type(8)))  float  v8f;
typedef int v4i_gcc __attribute__((vector_size(16)));  // matches builtin param

#define LDS_STRIDE 40  // halves; 80 bytes/row -> every 16B chunk stays aligned

#define AS1_ __attribute__((address_space(1)))
#define AS3_ __attribute__((address_space(3)))
typedef AS1_ v4i_gcc* gvec_ptr;
typedef AS3_ v4i_gcc* lvec_ptr;

#if defined(__AMDGCN__) && __has_builtin(__builtin_amdgcn_global_load_async_to_lds_b128)
#define HAS_ASYNC_LDS 1
#else
#define HAS_ASYNC_LDS 0
#endif

// 32-byte global -> LDS tile copy (pure copy path).
// Uses CDNA5 GLOBAL_LOAD_ASYNC_TO_LDS_B128 (ASYNCcnt) when available; otherwise
// a paired VGPR round-trip (both loads issued before stores so they clause).
__device__ __forceinline__ void cp32_to_lds(bf16* lds, const bf16* g) {
#if HAS_ASYNC_LDS
  __builtin_amdgcn_global_load_async_to_lds_b128((gvec_ptr)g, (lvec_ptr)lds, 0, 0);
  __builtin_amdgcn_global_load_async_to_lds_b128((gvec_ptr)(g + 8),
                                                 (lvec_ptr)(lds + 8), 0, 0);
#else
  v8bf x0 = *(const v8bf*)g;
  v8bf x1 = *(const v8bf*)(g + 8);
  *(v8bf*)lds = x0;
  *(v8bf*)(lds + 8) = x1;
#endif
}

__device__ __forceinline__ void cp16_to_lds(bf16* lds, const bf16* g) {
#if HAS_ASYNC_LDS
  __builtin_amdgcn_global_load_async_to_lds_b128((gvec_ptr)g, (lvec_ptr)lds, 0, 0);
#else
  *(v8bf*)lds = *(const v8bf*)g;
#endif
}

// barrier after staging: async copies must be retired (ASYNCcnt==0) before any
// wave may signal; ds stores are handled by the compiler's dscnt waits.
__device__ __forceinline__ void stage_fence() {
#if HAS_ASYNC_LDS
#if __has_builtin(__builtin_amdgcn_s_wait_asynccnt)
  __builtin_amdgcn_s_wait_asynccnt(0);
#else
  asm volatile("s_wait_asynccnt 0" ::: "memory");
#endif
#endif
  __syncthreads();
}

// ---------------- WMMA helpers ----------------
__device__ __forceinline__ v8f wmma_bf16(v16bf a, v16bf b, v8f c) {
  // (neg_a, A, neg_b, B, c_mod, C, reuse_a, reuse_b)
  return __builtin_amdgcn_wmma_f32_16x16x32_bf16(false, a, false, b, (short)0, c,
                                                 false, false);
}

__device__ __forceinline__ v16bf frag_cat(v8bf lo, v8bf hi) {
  v16bf r;
#pragma unroll
  for (int i = 0; i < 8; ++i) { r[i] = lo[i]; r[8 + i] = hi[i]; }
  return r;
}

// A fragment (16x32 bf16): lane<16 -> K 0-7 & 16-23 ; lane>=16 -> K 8-15 & 24-31
__device__ __forceinline__ v16bf load_frag_a(const bf16* rowm, int lane) {
  const int kb = (lane >= 16) ? 8 : 0;
  return frag_cat(*(const v8bf*)(rowm + kb), *(const v8bf*)(rowm + 16 + kb));
}

// B fragment from K-major LDS row ([n][k]): lane<16 -> K 0-7 & 16-23 ; lane>=16 -> K 8-15 & 24-31
__device__ __forceinline__ v16bf load_frag_bt(const bf16* rown, int lane) {
  const int kb = (lane < 16) ? 0 : 8;
  return frag_cat(*(const v8bf*)(rown + kb), *(const v8bf*)(rown + 16 + kb));
}

__device__ __forceinline__ float gelu_tanh(float x) {
  const float c = 0.79788456080286535588f; // sqrt(2/pi)
  return 0.5f * x * (1.0f + tanhf(c * (x + 0.044715f * x * x * x)));
}

// ---------------- generic GEMM: C = A(bf16, MxK) @ W(f32, KxN) + bias ----------------
// block = 256 threads (8 waves), block tile 128(M)x64(N), waves 4(M) x 2(N),
// each wave owns a 32x32 tile = 4 independent WMMA accumulator chains.
__global__ void __launch_bounds__(256) gemm_bf16_k(
    const bf16* __restrict__ A, int lda,
    const float* __restrict__ W, int ldw,
    const float* __restrict__ bias,
    float* __restrict__ outF, bf16* __restrict__ outB, int ldc,
    int Kdim, int act) {
  __shared__ __align__(16) bf16 As[128 * LDS_STRIDE];
  __shared__ __align__(16) bf16 Bs[64 * LDS_STRIDE];
  const int t = threadIdx.x, lane = t & 31, wave = t >> 5;
  const int wm = wave >> 1, wn = wave & 1;
  const int mblk = blockIdx.y * 128, nblk = blockIdx.x * 64;
  const int ar = t >> 1, ac = (t & 1) * 16;  // A stage: 128 rows x 32 k
  const int wk = t >> 3, wn8 = (t & 7) * 8;  // W stage: 32 k x 64 n (transpose)
  v8f c00 = {}, c01 = {}, c10 = {}, c11 = {};
  for (int kk = 0; kk < Kdim; kk += 32) {
    const bf16* ag = &A[(size_t)(mblk + ar) * lda + kk + ac];
    cp32_to_lds(&As[ar * LDS_STRIDE + ac], ag);
    const float* wp = &W[(size_t)(kk + wk) * ldw + nblk + wn8];
    float4 w0 = *(const float4*)wp;
    float4 w1 = *(const float4*)(wp + 4);
    Bs[(wn8 + 0) * LDS_STRIDE + wk] = (bf16)w0.x;
    Bs[(wn8 + 1) * LDS_STRIDE + wk] = (bf16)w0.y;
    Bs[(wn8 + 2) * LDS_STRIDE + wk] = (bf16)w0.z;
    Bs[(wn8 + 3) * LDS_STRIDE + wk] = (bf16)w0.w;
    Bs[(wn8 + 4) * LDS_STRIDE + wk] = (bf16)w1.x;
    Bs[(wn8 + 5) * LDS_STRIDE + wk] = (bf16)w1.y;
    Bs[(wn8 + 6) * LDS_STRIDE + wk] = (bf16)w1.z;
    Bs[(wn8 + 7) * LDS_STRIDE + wk] = (bf16)w1.w;
    if (kk + 32 < Kdim) {  // overlap next tile's HBM latency with this MAC phase
      __builtin_prefetch(ag + 32, 0, 1);
      __builtin_prefetch(wp + (size_t)32 * ldw, 0, 1);
    }
    stage_fence();
    v16bf a0 = load_frag_a(&As[(wm * 32 + (lane & 15)) * LDS_STRIDE], lane);
    v16bf a1 = load_frag_a(&As[(wm * 32 + 16 + (lane & 15)) * LDS_STRIDE], lane);
    v16bf b0 = load_frag_bt(&Bs[(wn * 32 + (lane & 15)) * LDS_STRIDE], lane);
    v16bf b1 = load_frag_bt(&Bs[(wn * 32 + 16 + (lane & 15)) * LDS_STRIDE], lane);
    c00 = wmma_bf16(a0, b0, c00);
    c01 = wmma_bf16(a0, b1, c01);
    c10 = wmma_bf16(a1, b0, c10);
    c11 = wmma_bf16(a1, b1, c11);
    __syncthreads();
  }
  const int ncol = nblk + wn * 32 + (lane & 15);
  const int mrow = mblk + wm * 32 + ((lane >> 4) << 3);
  const float bia0 = bias[ncol], bia1 = bias[ncol + 16];
#pragma unroll
  for (int r = 0; r < 8; ++r) {
    float v00 = c00[r] + bia0, v01 = c01[r] + bia1;
    float v10 = c10[r] + bia0, v11 = c11[r] + bia1;
    if (act) {
      v00 = gelu_tanh(v00); v01 = gelu_tanh(v01);
      v10 = gelu_tanh(v10); v11 = gelu_tanh(v11);
    }
    const size_t o0 = (size_t)(mrow + r) * ldc + ncol;
    const size_t o1 = (size_t)(mrow + 16 + r) * ldc + ncol;
    if (outF) { outF[o0] = v00; outF[o0 + 16] = v01;
                outF[o1] = v10; outF[o1 + 16] = v11; }
    if (outB) { outB[o0] = (bf16)v00; outB[o0 + 16] = (bf16)v01;
                outB[o1] = (bf16)v10; outB[o1 + 16] = (bf16)v11; }
  }
}

// ---------------- attention: S = Q @ K^T * (1/sqrt(DH)) ----------------
// block tile 128(M:q) x 64(N:k), grid (2, 1, 96)
__global__ void __launch_bounds__(256) attn_scores_k(
    const bf16* __restrict__ qkv, float* __restrict__ scores) {
  __shared__ __align__(16) bf16 As[128 * LDS_STRIDE];
  __shared__ __align__(16) bf16 Bs[64 * LDS_STRIDE];
  const int z = blockIdx.z, bb = z / NH_, hh = z % NH_;
  const bf16* Qb = qkv + (size_t)bb * T_ * QKV_ + hh * DH_;
  const bf16* Kb = qkv + (size_t)bb * T_ * QKV_ + H_ + hh * DH_;
  const int t = threadIdx.x, lane = t & 31, wave = t >> 5;
  const int wm = wave >> 1, wn = wave & 1;
  const int nblk = blockIdx.x * 64;
  const int ar = t >> 1, ac = (t & 1) * 16;  // Q stage: 128 rows x 32 k
  const int br = t >> 2, bc = (t & 3) * 8;   // K stage: 64 rows x 32 k
  v8f c00 = {}, c01 = {}, c10 = {}, c11 = {};
  for (int kk = 0; kk < DH_; kk += 32) {
    cp32_to_lds(&As[ar * LDS_STRIDE + ac], &Qb[(size_t)ar * QKV_ + kk + ac]);
    // K rows become B columns: rows of K are already the K-major [n][k] layout
    cp16_to_lds(&Bs[br * LDS_STRIDE + bc],
                &Kb[(size_t)(nblk + br) * QKV_ + kk + bc]);
    stage_fence();
    v16bf a0 = load_frag_a(&As[(wm * 32 + (lane & 15)) * LDS_STRIDE], lane);
    v16bf a1 = load_frag_a(&As[(wm * 32 + 16 + (lane & 15)) * LDS_STRIDE], lane);
    v16bf b0 = load_frag_bt(&Bs[(wn * 32 + (lane & 15)) * LDS_STRIDE], lane);
    v16bf b1 = load_frag_bt(&Bs[(wn * 32 + 16 + (lane & 15)) * LDS_STRIDE], lane);
    c00 = wmma_bf16(a0, b0, c00);
    c01 = wmma_bf16(a0, b1, c01);
    c10 = wmma_bf16(a1, b0, c10);
    c11 = wmma_bf16(a1, b1, c11);
    __syncthreads();
  }
  float* Sb = scores + (size_t)z * T_ * T_;
  const int ncol = nblk + wn * 32 + (lane & 15);
  const int mrow = wm * 32 + ((lane >> 4) << 3);
  const float scale = 0.125f; // 1/sqrt(64)
#pragma unroll
  for (int r = 0; r < 8; ++r) {
    Sb[(size_t)(mrow + r) * T_ + ncol]           = c00[r] * scale;
    Sb[(size_t)(mrow + r) * T_ + ncol + 16]      = c01[r] * scale;
    Sb[(size_t)(mrow + 16 + r) * T_ + ncol]      = c10[r] * scale;
    Sb[(size_t)(mrow + 16 + r) * T_ + ncol + 16] = c11[r] * scale;
  }
}

// ---------------- softmax over last dim (128) with mask bias, -> bf16 P ----------------
__global__ void __launch_bounds__(128) softmax_k(
    const float* __restrict__ s, const int* __restrict__ mask, bf16* __restrict__ p) {
  const int q = blockIdx.x, z = blockIdx.y, bb = z / NH_;
  const int t = threadIdx.x;
  __shared__ float red[T_];
  const size_t base = ((size_t)z * T_ + q) * T_;
  const float bias = (1.0f - (float)mask[bb * T_ + t]) * -10000.0f;
  const float v = s[base + t] + bias;
  red[t] = v; __syncthreads();
  for (int st = 64; st > 0; st >>= 1) {
    if (t < st) red[t] = fmaxf(red[t], red[t + st]);
    __syncthreads();
  }
  const float m = red[0]; __syncthreads();
  const float e = __expf(v - m);
  red[t] = e; __syncthreads();
  for (int st = 64; st > 0; st >>= 1) {
    if (t < st) red[t] += red[t + st];
    __syncthreads();
  }
  p[base + t] = (bf16)(e / red[0]);
}

// ---------------- ctx = P @ V  (128x128 @ 128x64 per head), grid (1, 96) ----------------
__global__ void __launch_bounds__(256) attn_ctx_k(
    const bf16* __restrict__ p, const bf16* __restrict__ qkv, bf16* __restrict__ ctx) {
  __shared__ __align__(16) bf16 As[128 * LDS_STRIDE];
  __shared__ __align__(16) bf16 Bs[64 * LDS_STRIDE];
  const int z = blockIdx.y, bb = z / NH_, hh = z % NH_;
  const bf16* Pb = p + (size_t)z * T_ * T_;
  const bf16* Vb = qkv + (size_t)bb * T_ * QKV_ + 2 * H_ + hh * DH_;
  const int t = threadIdx.x, lane = t & 31, wave = t >> 5;
  const int wm = wave >> 1, wn = wave & 1;
  const int ar = t >> 1, ac = (t & 1) * 16;  // P stage: 128 rows x 32 k
  const int wk = t >> 3, wn8 = (t & 7) * 8;  // V stage: 32 k x 64 n (transpose)
  v8f c00 = {}, c01 = {}, c10 = {}, c11 = {};
  for (int kk = 0; kk < T_; kk += 32) {
    cp32_to_lds(&As[ar * LDS_STRIDE + ac], &Pb[(size_t)ar * T_ + kk + ac]);
    v8bf vv = *(const v8bf*)&Vb[(size_t)(kk + wk) * QKV_ + wn8];
#pragma unroll
    for (int i = 0; i < 8; ++i) Bs[(wn8 + i) * LDS_STRIDE + wk] = vv[i];
    stage_fence();
    v16bf a0 = load_frag_a(&As[(wm * 32 + (lane & 15)) * LDS_STRIDE], lane);
    v16bf a1 = load_frag_a(&As[(wm * 32 + 16 + (lane & 15)) * LDS_STRIDE], lane);
    v16bf b0 = load_frag_bt(&Bs[(wn * 32 + (lane & 15)) * LDS_STRIDE], lane);
    v16bf b1 = load_frag_bt(&Bs[(wn * 32 + 16 + (lane & 15)) * LDS_STRIDE], lane);
    c00 = wmma_bf16(a0, b0, c00);
    c01 = wmma_bf16(a0, b1, c01);
    c10 = wmma_bf16(a1, b0, c10);
    c11 = wmma_bf16(a1, b1, c11);
    __syncthreads();
  }
  const int ncol = wn * 32 + (lane & 15);
  const int mrow = wm * 32 + ((lane >> 4) << 3);
#pragma unroll
  for (int r = 0; r < 8; ++r) {
    bf16* o0 = &ctx[(size_t)(bb * T_ + mrow + r) * H_ + hh * DH_ + ncol];
    bf16* o1 = &ctx[(size_t)(bb * T_ + mrow + 16 + r) * H_ + hh * DH_ + ncol];
    o0[0]  = (bf16)c00[r]; o0[16] = (bf16)c01[r];
    o1[0]  = (bf16)c10[r]; o1[16] = (bf16)c11[r];
  }
}

// ---------------- embeddings + LayerNorm ----------------
__global__ void __launch_bounds__(256) emb_ln_k(
    const int* __restrict__ sent, const float* __restrict__ wemb,
    const float* __restrict__ pemb, const float* __restrict__ temb,
    const float* __restrict__ g, const float* __restrict__ be,
    float* __restrict__ outF, bf16* __restrict__ outB) {
  const int row = blockIdx.x, t = threadIdx.x;
  const int tok = sent[row], pos = row % T_;
  __shared__ float s1[256], s2[256];
  float x[3], sum = 0.f, sq = 0.f;
#pragma unroll
  for (int i = 0; i < 3; ++i) {
    const int j = t + i * 256;
    const float v = wemb[(size_t)tok * H_ + j] + pemb[(size_t)pos * H_ + j] + temb[j];
    x[i] = v; sum += v; sq += v * v;
  }
  s1[t] = sum; s2[t] = sq; __syncthreads();
  for (int st = 128; st > 0; st >>= 1) {
    if (t < st) { s1[t] += s1[t + st]; s2[t] += s2[t + st]; }
    __syncthreads();
  }
  const float mean = s1[0] * (1.0f / H_);
  const float var  = s2[0] * (1.0f / H_) - mean * mean;
  const float rstd = rsqrtf(var + 1e-12f);
#pragma unroll
  for (int i = 0; i < 3; ++i) {
    const int j = t + i * 256;
    const float y = (x[i] - mean) * rstd * g[j] + be[j];
    outF[(size_t)row * H_ + j] = y;
    outB[(size_t)row * H_ + j] = (bf16)y;
  }
}

// ---------------- residual + LayerNorm ----------------
__global__ void __launch_bounds__(256) ln_res_k(
    const float* __restrict__ resid, const float* __restrict__ delta,
    const float* __restrict__ g, const float* __restrict__ be,
    float* __restrict__ outF, bf16* __restrict__ outB) {
  const int row = blockIdx.x, t = threadIdx.x;
  __shared__ float s1[256], s2[256];
  float x[3], sum = 0.f, sq = 0.f;
#pragma unroll
  for (int i = 0; i < 3; ++i) {
    const int j = t + i * 256;
    const float v = resid[(size_t)row * H_ + j] + delta[(size_t)row * H_ + j];
    x[i] = v; sum += v; sq += v * v;
  }
  s1[t] = sum; s2[t] = sq; __syncthreads();
  for (int st = 128; st > 0; st >>= 1) {
    if (t < st) { s1[t] += s1[t + st]; s2[t] += s2[t + st]; }
    __syncthreads();
  }
  const float mean = s1[0] * (1.0f / H_);
  const float var  = s2[0] * (1.0f / H_) - mean * mean;
  const float rstd = rsqrtf(var + 1e-12f);
#pragma unroll
  for (int i = 0; i < 3; ++i) {
    const int j = t + i * 256;
    const float y = (x[i] - mean) * rstd * g[j] + be[j];
    outF[(size_t)row * H_ + j] = y;
    outB[(size_t)row * H_ + j] = (bf16)y;
  }
}

// ---------------- tag projection: feats = h @ Wt + bt (H x 5) ----------------
__global__ void __launch_bounds__(256) tagproj_k(
    const float* __restrict__ h, const float* __restrict__ Wt,
    const float* __restrict__ bt, float* __restrict__ feats) {
  const int row = blockIdx.x, t = threadIdx.x;
  __shared__ float red[256];
  float loc[KT] = {};
  for (int j = t; j < H_; j += 256) {
    const float a = h[(size_t)row * H_ + j];
#pragma unroll
    for (int k = 0; k < KT; ++k) loc[k] += a * Wt[j * KT + k];
  }
  for (int k = 0; k < KT; ++k) {
    red[t] = loc[k]; __syncthreads();
    for (int st = 128; st > 0; st >>= 1) {
      if (t < st) red[t] += red[t + st];
      __syncthreads();
    }
    if (t == 0) feats[(size_t)row * KT + k] = red[0] + bt[k];
    __syncthreads();
  }
}

// ---------------- CRF Viterbi decode (one thread per batch) ----------------
__global__ void viterbi_k(const float* __restrict__ feats,
                          const float* __restrict__ trans,
                          float* __restrict__ out) {
  __shared__ int psi[B_][T_ - 1][KT];
  const int b = threadIdx.x;
  if (b >= B_) return;
  float tr[KT * KT];
#pragma unroll
  for (int i = 0; i < KT * KT; ++i) tr[i] = trans[i];
  float ld[KT];
#pragma unroll
  for (int k = 0; k < KT; ++k) ld[k] = (k == STARTTAG) ? 0.0f : -10000.0f;
  for (int tt = 1; tt < T_; ++tt) {
    float nld[KT];
#pragma unroll
    for (int to = 0; to < KT; ++to) {
      float best = tr[to * KT] + ld[0];
      int arg = 0;
#pragma unroll
      for (int fr = 1; fr < KT; ++fr) {
        const float v = tr[to * KT + fr] + ld[fr];
        if (v > best) { best = v; arg = fr; }
      }
      psi[b][tt - 1][to] = arg;
      nld[to] = best + feats[((size_t)b * T_ + tt) * KT + to];
    }
#pragma unroll
    for (int k = 0; k < KT; ++k) ld[k] = nld[k];
  }
  float sc = ld[0]; int last = 0;
#pragma unroll
  for (int k = 1; k < KT; ++k)
    if (ld[k] > sc) { sc = ld[k]; last = k; }
  out[b] = sc;
  out[B_ + (size_t)b * T_ + (T_ - 1)] = (float)last;
  int pc = last;
  for (int tt = T_ - 2; tt >= 0; --tt) {
    pc = psi[b][tt][pc];
    out[B_ + (size_t)b * T_ + tt] = (float)pc;
  }
}

// ---------------- host orchestration ----------------
extern "C" void kernel_launch(void* const* d_in, const int* in_sizes, int n_in,
                              void* d_out, int out_size, void* d_ws, size_t ws_size,
                              hipStream_t stream) {
  (void)in_sizes; (void)n_in; (void)out_size; (void)ws_size;
  const int*   sent  = (const int*)d_in[0];
  const int*   mask  = (const int*)d_in[1];
  const float* wemb  = (const float*)d_in[2];
  const float* pemb  = (const float*)d_in[3];
  const float* temb  = (const float*)d_in[4];
  const float* elns  = (const float*)d_in[5];
  const float* elnb  = (const float*)d_in[6];
  const float* Wqkv  = (const float*)d_in[7];
  const float* bqkv  = (const float*)d_in[8];
  const float* Wo    = (const float*)d_in[9];
  const float* bo    = (const float*)d_in[10];
  const float* ln1s  = (const float*)d_in[11];
  const float* ln1b  = (const float*)d_in[12];
  const float* W1    = (const float*)d_in[13];
  const float* b1    = (const float*)d_in[14];
  const float* W2    = (const float*)d_in[15];
  const float* b2    = (const float*)d_in[16];
  const float* ln2s  = (const float*)d_in[17];
  const float* ln2b  = (const float*)d_in[18];
  const float* Wt    = (const float*)d_in[19];
  const float* bt    = (const float*)d_in[20];
  const float* trans = (const float*)d_in[21];

  char* ws = (char*)d_ws;
  size_t off = 0;
  auto alloc = [&](size_t bytes) -> char* {
    char* p = ws + off;
    off = (off + bytes + 255) & ~(size_t)255;
    return p;
  };
  float* hF      = (float*)alloc((size_t)NTOK * H_ * 4);
  bf16*  hB      = (bf16*) alloc((size_t)NTOK * H_ * 2);
  bf16*  qkvB    = (bf16*) alloc((size_t)NTOK * QKV_ * 2);
  float* scoresF = (float*)alloc((size_t)B_ * NH_ * T_ * T_ * 4);
  bf16*  pB      = (bf16*) alloc((size_t)B_ * NH_ * T_ * T_ * 2);
  bf16*  ctxB    = (bf16*) alloc((size_t)NTOK * H_ * 2);
  float* deltaF  = (float*)alloc((size_t)NTOK * H_ * 4);
  bf16*  ff1B    = (bf16*) alloc((size_t)NTOK * F_ * 2);
  float* featsF  = (float*)alloc((size_t)NTOK * KT * 4);

  emb_ln_k<<<NTOK, 256, 0, stream>>>(sent, wemb, pemb, temb, elns, elnb, hF, hB);

  for (int l = 0; l < L_; ++l) {
    // QKV projection -> bf16 [1024 x 2304]
    gemm_bf16_k<<<dim3(QKV_ / 64, NTOK / 128), 256, 0, stream>>>(
        hB, H_, Wqkv + (size_t)l * H_ * QKV_, QKV_, bqkv + (size_t)l * QKV_,
        nullptr, qkvB, QKV_, H_, 0);
    // S = Q K^T / sqrt(dh)
    attn_scores_k<<<dim3(T_ / 64, 1, B_ * NH_), 256, 0, stream>>>(qkvB, scoresF);
    // softmax (+mask bias) -> bf16 P
    softmax_k<<<dim3(T_, B_ * NH_), T_, 0, stream>>>(scoresF, mask, pB);
    // ctx = P V -> bf16 [1024 x 768] (heads interleaved back into H)
    attn_ctx_k<<<dim3(1, B_ * NH_), 256, 0, stream>>>(pB, qkvB, ctxB);
    // output projection -> fp32 delta
    gemm_bf16_k<<<dim3(H_ / 64, NTOK / 128), 256, 0, stream>>>(
        ctxB, H_, Wo + (size_t)l * H_ * H_, H_, bo + (size_t)l * H_,
        deltaF, nullptr, H_, H_, 0);
    ln_res_k<<<NTOK, 256, 0, stream>>>(hF, deltaF, ln1s + (size_t)l * H_,
                                       ln1b + (size_t)l * H_, hF, hB);
    // FFN1 + GELU -> bf16 [1024 x 3072]
    gemm_bf16_k<<<dim3(F_ / 64, NTOK / 128), 256, 0, stream>>>(
        hB, H_, W1 + (size_t)l * H_ * F_, F_, b1 + (size_t)l * F_,
        nullptr, ff1B, F_, H_, 1);
    // FFN2 -> fp32 delta
    gemm_bf16_k<<<dim3(H_ / 64, NTOK / 128), 256, 0, stream>>>(
        ff1B, F_, W2 + (size_t)l * F_ * H_, H_, b2 + (size_t)l * H_,
        deltaF, nullptr, H_, F_, 0);
    ln_res_k<<<NTOK, 256, 0, stream>>>(hF, deltaF, ln2s + (size_t)l * H_,
                                       ln2b + (size_t)l * H_, hF, hB);
  }

  tagproj_k<<<NTOK, 256, 0, stream>>>(hF, Wt, bt, featsF);
  viterbi_k<<<1, 32, 0, stream>>>(featsF, trans, (float*)d_out);
}
